// DependencyParser_22351009809139
// MI455X (gfx1250) — compile-verified, matrix operand order checked by reading, not run on hardware
//
#include <hip/hip_runtime.h>
#include <hip/hip_bf16.h>

#define H     300
#define MID   100
#define NS    1024
#define NP1   1025

typedef __attribute__((ext_vector_type(16))) __bf16    v16bf;
typedef __attribute__((ext_vector_type(8)))  float     v8f;
typedef __attribute__((ext_vector_type(8)))  unsigned  v8u;

// Pack two f32 into one dword of two bf16 (truncating round; plenty for bf16 anyway).
__device__ __forceinline__ unsigned pk(float lo, float hi) {
  const unsigned ul = __builtin_bit_cast(unsigned, lo);
  const unsigned uh = __builtin_bit_cast(unsigned, hi);
  return (ul >> 16) | (uh & 0xFFFF0000u);
}

__device__ __forceinline__ v8u pack8(float4 a, float4 b, float4 c, float4 d) {
  v8u r;
  r[0] = pk(a.x, a.y); r[1] = pk(a.z, a.w);
  r[2] = pk(b.x, b.y); r[3] = pk(b.z, b.w);
  r[4] = pk(c.x, c.y); r[5] = pk(c.z, c.w);
  r[6] = pk(d.x, d.y); r[7] = pk(d.z, d.w);
  return r;
}

// Fused GEMM: [1024x300] x [300x200] -> cols 0..99 = s1+b1, cols 100..199 = s2 rows 1..1024.
// One wave per 16x16 output tile. 9 branch-free K=32 steps (float4 loads) + 1 masked tail.
// A 16x32 bf16 fragment layout: elem e of lane -> K = ((e&8)<<1) + (lane>=16 ? 8:0) + (e&7),
// i.e. per lane two contiguous 8-float runs at K0+koff and K0+16+koff.
__global__ void gemm_s1s2(const float* __restrict__ sentence,
                          const float* __restrict__ W1,
                          const float* __restrict__ b1,
                          float* __restrict__ s1b,
                          float* __restrict__ s2) {
  const int lane = threadIdx.x;        // 0..31
  const int mt   = blockIdx.x;         // 0..63
  const int nt   = blockIdx.y;         // 0..12 (N padded 200 -> 208)
  const int rsel = lane >> 4;          // lane half
  const int lr   = lane & 15;
  const int koff = rsel << 3;          // 0 or 8
  const int rowA = mt * 16 + lr;       // < 1024 always
  const int colB = nt * 16 + lr;       // output column (may be >= 200: discarded at store)
  const int cb   = (colB < 2 * MID) ? colB : (2 * MID - 1);   // clamp for safe loads
  const int brow = (cb < MID) ? cb : (cb - MID);
  const int boff = (cb < MID) ? 0 : H;

  const float* __restrict__ arow = sentence + rowA * H;
  const float* __restrict__ bsrc = W1 + brow * (2 * H) + boff;

  v8f c = {0.f, 0.f, 0.f, 0.f, 0.f, 0.f, 0.f, 0.f};

#pragma unroll
  for (int step = 0; step < 9; ++step) {          // K = 0..287, fully in-range
    const int kc = step * 32 + koff;
    const float4 a0 = *(const float4*)(arow + kc);
    const float4 a1 = *(const float4*)(arow + kc + 4);
    const float4 a2 = *(const float4*)(arow + kc + 16);
    const float4 a3 = *(const float4*)(arow + kc + 20);
    const float4 b0 = *(const float4*)(bsrc + kc);
    const float4 b1v = *(const float4*)(bsrc + kc + 4);
    const float4 b2 = *(const float4*)(bsrc + kc + 16);
    const float4 b3 = *(const float4*)(bsrc + kc + 20);
    const v16bf a = __builtin_bit_cast(v16bf, pack8(a0, a1, a2, a3));
    const v16bf b = __builtin_bit_cast(v16bf, pack8(b0, b1v, b2, b3));
    c = __builtin_amdgcn_wmma_f32_16x16x32_bf16(false, a, false, b,
                                                (short)0, c, false, false);
  }
  {                                               // K tail: 288..319, valid < 300
    v8u ap, bp;
#pragma unroll
    for (int w = 0; w < 8; ++w) {
      const int e0 = 2 * w;
      const int k0 = 288 + ((e0 & 8) << 1) + koff + (e0 & 7);
      const int k1 = k0 + 1;
      const int k0c = (k0 < H) ? k0 : 0;
      const int k1c = (k1 < H) ? k1 : 0;
      const float m0 = (k0 < H) ? 1.f : 0.f;
      const float m1 = (k1 < H) ? 1.f : 0.f;
      ap[w] = pk(arow[k0c] * m0, arow[k1c] * m1);
      bp[w] = pk(bsrc[k0c] * m0, bsrc[k1c] * m1);
    }
    c = __builtin_amdgcn_wmma_f32_16x16x32_bf16(false, __builtin_bit_cast(v16bf, ap),
                                                false, __builtin_bit_cast(v16bf, bp),
                                                (short)0, c, false, false);
  }

#pragma unroll
  for (int d = 0; d < 8; ++d) {
    const int om = mt * 16 + d + (rsel << 3);     // C/D layout: M = d + 8*(lane>=16)
    const int on = colB;
    const float v = c[d];
    if (on < MID) {
      s1b[om * MID + on] = v + b1[on];            // fold b1 into s1
    } else if (on < 2 * MID) {
      s2[(om + 1) * MID + (on - MID)] = v;        // s2 rows 1..1024; row 0 is implicit zero
    }
  }
}

// Dominant kernel: 16(i) x 16(j) pair tile per 256-thread block; float4 LDS reads,
// sigmoid via v_exp_f32 + v_rcp_f32, reduction over MID with W2.
__global__ void pair_scores(const float* __restrict__ s1b,
                            const float* __restrict__ s2,
                            const float* __restrict__ W2,
                            const float* __restrict__ b2,
                            float* __restrict__ out_scores) {
  __shared__ __align__(16) float s1s[16][MID];
  __shared__ __align__(16) float s2s[16][MID];
  __shared__ __align__(16) float w2s[MID];
  const int tid = threadIdx.x;
  const int it  = blockIdx.y;   // 0..63
  const int jt  = blockIdx.x;   // 0..64

  for (int idx = tid; idx < 16 * MID; idx += 256) {
    const int r = idx / MID, m = idx % MID;
    s1s[r][m] = s1b[(it * 16 + r) * MID + m];
    const int j = jt * 16 + r;
    s2s[r][m] = (j == 0 || j >= NP1) ? 0.f : s2[j * MID + m];  // j==0: zero head row
  }
  if (tid < MID) w2s[tid] = W2[tid];
  __syncthreads();

  const int ti = tid >> 4, tj = tid & 15;
  const int i = it * 16 + ti;
  const int j = jt * 16 + tj;

  const float4* __restrict__ xv = (const float4*)&s1s[ti][0];
  const float4* __restrict__ yv = (const float4*)&s2s[tj][0];
  const float4* __restrict__ wv = (const float4*)&w2s[0];

  float acc = 0.f;
#pragma unroll 5
  for (int q = 0; q < MID / 4; ++q) {
    const float4 x = xv[q], y = yv[q], w = wv[q];
    acc = fmaf(w.x, __builtin_amdgcn_rcpf(1.f + __expf(-(x.x + y.x))), acc);
    acc = fmaf(w.y, __builtin_amdgcn_rcpf(1.f + __expf(-(x.y + y.y))), acc);
    acc = fmaf(w.z, __builtin_amdgcn_rcpf(1.f + __expf(-(x.z + y.z))), acc);
    acc = fmaf(w.w, __builtin_amdgcn_rcpf(1.f + __expf(-(x.w + y.w))), acc);
  }
  if (j < NP1) out_scores[i * NP1 + j] = acc + b2[0];
}

// Per-row: L1-loss partial against one-hot target + in-place softmax over 1025 columns.
__global__ void softmax_loss_rows(float* __restrict__ scores,
                                  const int* __restrict__ tgt,
                                  float* __restrict__ rowloss) {
  __shared__ float red[256];
  const int tid = threadIdx.x;
  const int i   = blockIdx.x;
  const int t   = tgt[i];
  float v[5];
  float mx = -3.4e38f, ls = 0.f;
#pragma unroll
  for (int k = 0; k < 5; ++k) {
    const int j = tid + (k << 8);
    float x = -3.4e38f;
    if (j < NP1) {
      x = scores[i * NP1 + j];
      const float ind = (j == t) ? 1.f : 0.f;
      ls += fabsf(x - ind);
    }
    v[k] = x;
    mx = fmaxf(mx, x);
  }
  red[tid] = mx; __syncthreads();
  for (int s = 128; s > 0; s >>= 1) { if (tid < s) red[tid] = fmaxf(red[tid], red[tid + s]); __syncthreads(); }
  const float rowmax = red[0]; __syncthreads();

  red[tid] = ls; __syncthreads();
  for (int s = 128; s > 0; s >>= 1) { if (tid < s) red[tid] += red[tid + s]; __syncthreads(); }
  if (tid == 0) rowloss[i] = red[0];
  __syncthreads();

  float es = 0.f;
#pragma unroll
  for (int k = 0; k < 5; ++k) {
    const int j = tid + (k << 8);
    if (j < NP1) { v[k] = __expf(v[k] - rowmax); es += v[k]; }
  }
  red[tid] = es; __syncthreads();
  for (int s = 128; s > 0; s >>= 1) { if (tid < s) red[tid] += red[tid + s]; __syncthreads(); }
  const float inv = __builtin_amdgcn_rcpf(red[0]);
#pragma unroll
  for (int k = 0; k < 5; ++k) {
    const int j = tid + (k << 8);
    if (j < NP1) scores[i * NP1 + j] = v[k] * inv;
  }
}

__global__ void finish_loss(const float* __restrict__ rowloss, float* __restrict__ out0) {
  __shared__ float red[256];
  const int tid = threadIdx.x;
  float s = 0.f;
  for (int r = tid; r < NS; r += 256) s += rowloss[r];
  red[tid] = s; __syncthreads();
  for (int k = 128; k > 0; k >>= 1) { if (tid < k) red[tid] += red[tid + k]; __syncthreads(); }
  if (tid == 0) out0[0] = red[0] * (1.f / (float)(NS * NP1));
}

extern "C" void kernel_launch(void* const* d_in, const int* in_sizes, int n_in,
                              void* d_out, int out_size, void* d_ws, size_t ws_size,
                              hipStream_t stream) {
  const float* sentence = (const float*)d_in[0];
  const int*   tgt      = (const int*)  d_in[1];
  const float* W1       = (const float*)d_in[2];
  const float* b1       = (const float*)d_in[3];
  const float* W2       = (const float*)d_in[4];
  const float* b2       = (const float*)d_in[5];
  float* out = (float*)d_out;
  float* ws  = (float*)d_ws;

  float* s1b     = ws;                 // 1024*100
  float* s2      = s1b + NS * MID;     // 1025*100 (row 0 never read)
  float* rowloss = s2 + NP1 * MID;     // 1024
  float* scores  = out + 1;            // scores live in d_out, softmaxed in place

  gemm_s1s2        <<<dim3(64, 13),  32, 0, stream>>>(sentence, W1, b1, s1b, s2);
  pair_scores      <<<dim3(65, 64), 256, 0, stream>>>(s1b, s2, W2, b2, scores);
  softmax_loss_rows<<<dim3(NS),     256, 0, stream>>>(scores, tgt, rowloss);
  finish_loss      <<<1,            256, 0, stream>>>(rowloss, out);
}
